// BertSelfAttention0_32521492365987
// MI455X (gfx1250) — matrix operand order repaired and checked by reading, not compiled, not run on hardware
//
#include <hip/hip_runtime.h>
#include <hip/hip_bf16.h>

typedef _Float16 v16h __attribute__((ext_vector_type(16)));
typedef _Float16 v8h  __attribute__((ext_vector_type(8)));
typedef _Float16 v4h  __attribute__((ext_vector_type(4)));
typedef float    v8f  __attribute__((ext_vector_type(8)));
typedef unsigned int u32x4 __attribute__((ext_vector_type(4)));
typedef unsigned int u32x8 __attribute__((ext_vector_type(8)));

#define BB 4
#define SS 2048
#define HH 768
#define NHH 12
#define DD 64
#define SCALE 0.125f

static __device__ inline v16h mk16(v8h lo, v8h hi) {
    return __builtin_shufflevector(lo, hi, 0,1,2,3,4,5,6,7,8,9,10,11,12,13,14,15);
}

static __device__ inline v8f wmma_f16(v16h a, v16h b, v8f c) {
    return __builtin_amdgcn_wmma_f32_16x16x32_f16(false, a, false, b, (short)0, c, false, false);
}

// ---------------------------------------------------------------------------
// Tensor Data Mover: 2D tile load (global -> LDS), f16 elements.
// D# built per CDNA5 ISA; groups live in SGPRs
// (all inputs block-uniform).  2-group form: tensors up to 2D.
// ---------------------------------------------------------------------------
static __device__ inline void tdm_load_2d(unsigned int lds_off, const void* gptr,
                                          unsigned int tensor_d0, unsigned int tensor_d1,
                                          unsigned int tile_d0, unsigned int tile_d1,
                                          unsigned long long stride0_elems)
{
    unsigned long long ga = (unsigned long long)(size_t)gptr;
    u32x4 g0;
    g0[0] = 1u;                                            // count=1, no gather
    g0[1] = lds_off;                                       // LDS byte address
    g0[2] = (unsigned int)ga;                              // global_addr[31:0]
    g0[3] = (unsigned int)((ga >> 32) & 0x01FFFFFFu)       // global_addr[56:32]
          | (2u << 30);                                    // type=2 ("image")
    u32x8 g1;
    g1[0] = 1u << 16;                                      // data_size=1 -> 2 bytes
    g1[1] = (tensor_d0 & 0xFFFFu) << 16;                   // tensor_dim0[15:0]
    g1[2] = (tensor_d0 >> 16) | ((tensor_d1 & 0xFFFFu) << 16);
    g1[3] = (tensor_d1 >> 16) | (tile_d0 << 16);           // tile_dim0
    g1[4] = tile_d1;                                       // tile_dim1, tile_dim2=0
    g1[5] = (unsigned int)stride0_elems;                   // tensor_dim0_stride lo
    g1[6] = (unsigned int)(stride0_elems >> 32) & 0xFFFFu; // stride hi, stride1=0
    g1[7] = 0u;
    asm volatile("tensor_load_to_lds %0, %1" :: "s"(g0), "s"(g1) : "memory");
}

// ---------------------------------------------------------------------------
// Kernel 1: QKV projection.  grid = (S/128, NH, B), block = 256 (8 waves).
// Q,K stored [B][NH][S][D] f16 (Q pre-scaled, bias added).  V stored
// transposed [B][NH][D][S] f16 so attention PV B-frags are contiguous.
// ---------------------------------------------------------------------------
__global__ __launch_bounds__(256) void qkv_kernel(
    const float* __restrict__ e1, const float* __restrict__ e2,
    const float* __restrict__ e3,
    const float* __restrict__ Wq, const float* __restrict__ bq,
    const float* __restrict__ Wk, const float* __restrict__ bk,
    const float* __restrict__ Wv, const float* __restrict__ bv,
    _Float16* __restrict__ Qh, _Float16* __restrict__ Kh,
    _Float16* __restrict__ Vh)
{
    __shared__ _Float16 lds_x[128 * 32];      // [row][k]  8KB
    __shared__ _Float16 lds_w[3][64 * 32];    // [d][k] transposed, 3x4KB

    const int tid = threadIdx.x;
    const int s0  = blockIdx.x * 128;
    const int h   = blockIdx.y;
    const int b   = blockIdx.z;
    const int g   = h >> 2;                   // modality group
    const float* x = (g == 0) ? e1 : (g == 1) ? e2 : e3;

    const int wid  = tid >> 5;
    const int lane = tid & 31;
    const int l16  = lane & 15;
    const int koA  = (lane < 16) ? 0 : 8;     // A-frag lane K base
    const int koB  = (lane < 16) ? 0 : 16;    // B-frag lane K base

    v8f acc[3][4];
    #pragma unroll
    for (int m = 0; m < 3; ++m)
        #pragma unroll
        for (int nb = 0; nb < 4; ++nb)
            #pragma unroll
            for (int r = 0; r < 8; ++r) acc[m][nb][r] = 0.0f;

    const float* wptr[3] = { Wq + (size_t)h * HH * DD,
                             Wk + (size_t)h * HH * DD,
                             Wv + (size_t)h * HH * DD };

    for (int k0 = 0; k0 < HH; k0 += 32) {
        __syncthreads();
        // stage X chunk [128 x 32] -> LDS (f32 -> f16, packed b64 stores)
        const float* xbase = x + ((size_t)b * SS + s0) * HH + k0;
        #pragma unroll
        for (int it = 0; it < 4; ++it) {
            int idx = tid + 256 * it;             // 1024 float4 chunks
            int row = idx >> 3, c = idx & 7;
            float4 v = *(const float4*)(xbase + (size_t)row * HH + c * 4);
            v4h hv;
            hv[0] = (_Float16)v.x; hv[1] = (_Float16)v.y;
            hv[2] = (_Float16)v.z; hv[3] = (_Float16)v.w;
            *(v4h*)&lds_x[row * 32 + c * 4] = hv;
        }
        // warm L2 with next X chunk (one cacheline per two threads)
        if (k0 + 32 < HH)
            __builtin_prefetch(xbase + 32 + (size_t)(tid >> 1) * HH, 0, 1);
        // stage W chunks [32 x 64] transposed -> LDS [64][32]
        #pragma unroll
        for (int m = 0; m < 3; ++m) {
            const float* wbase = wptr[m] + (size_t)k0 * DD;
            #pragma unroll
            for (int it = 0; it < 8; ++it) {
                int idx = tid + 256 * it;         // 2048 elems
                int k = idx >> 6, d = idx & 63;
                lds_w[m][d * 32 + k] = (_Float16)wbase[k * DD + d];
            }
        }
        __syncthreads();

        // A fragment (shared by all 12 WMMAs this step)
        const int arow = wid * 16 + l16;
        v8h alo = *(const v8h*)&lds_x[arow * 32 + koA];
        v8h ahi = *(const v8h*)&lds_x[arow * 32 + 16 + koA];
        v16h a = mk16(alo, ahi);

        #pragma unroll
        for (int m = 0; m < 3; ++m) {
            v16h bf[4];
            #pragma unroll
            for (int nb = 0; nb < 4; ++nb) {
                int n = nb * 16 + l16;
                v8h blo = *(const v8h*)&lds_w[m][n * 32 + koA];
                v8h bhi = *(const v8h*)&lds_w[m][n * 32 + 16 + koA];
                bf[nb] = mk16(blo, bhi);
            }
            #pragma unroll
            for (int nb = 0; nb < 4; ++nb)
                acc[m][nb] = wmma_f16(a, bf[nb], acc[m][nb]);
        }
    }

    // epilogue: bias (+scale for Q), store f16
    const int hsel = (lane < 16) ? 0 : 8;
    #pragma unroll
    for (int nb = 0; nb < 4; ++nb) {
        int col = nb * 16 + l16;
        float bqv = bq[h * DD + col];
        float bkv = bk[h * DD + col];
        float bvv = bv[h * DD + col];
        #pragma unroll
        for (int r = 0; r < 8; ++r) {
            int srow = s0 + wid * 16 + r + hsel;
            size_t qo = (((size_t)b * NHH + h) * SS + srow) * DD + col;
            Qh[qo] = (_Float16)((acc[0][nb][r] + bqv) * SCALE);
            Kh[qo] = (_Float16)(acc[1][nb][r] + bkv);
            Vh[(((size_t)b * NHH + h) * DD + col) * SS + srow] =
                (_Float16)(acc[2][nb][r] + bvv);
        }
    }
}

// ---------------------------------------------------------------------------
// Kernel 2: flash attention.  grid = (S/128, NH, B), block = 256 (8 waves).
// K/V tiles are double-buffered and staged by the Tensor Data Mover: wave 0
// issues tile j+1's descriptors into the alternate buffer, then waits
// TENSORcnt<=2 (TDM in-order per wave => tile j complete, j+1 in flight),
// overlapping DMA with WMMA compute.
// ---------------------------------------------------------------------------
__global__ __launch_bounds__(256) void attn_kernel(
    const _Float16* __restrict__ Qh, const _Float16* __restrict__ Kh,
    const _Float16* __restrict__ Vh, float* __restrict__ out)
{
    __shared__ _Float16 sQ[128 * 64];      // 16KB
    __shared__ _Float16 sK[2][64 * 64];    // 2x8KB  [key][d]
    __shared__ _Float16 sV[2][64 * 64];    // 2x8KB  [d][key]
    __shared__ _Float16 sP[128 * 64];      // 16KB

    const int tid = threadIdx.x;
    const int s0  = blockIdx.x * 128;
    const int h   = blockIdx.y;
    const int b   = blockIdx.z;
    const size_t bh = (size_t)b * NHH + h;

    const _Float16* Qg = Qh + bh * SS * DD + (size_t)s0 * DD;
    const _Float16* Kg = Kh + bh * SS * DD;
    const _Float16* Vg = Vh + bh * DD * SS;   // [D][S]

    const int wid  = tid >> 5;
    const int lane = tid & 31;
    const int l16  = lane & 15;
    const int koA  = (lane < 16) ? 0 : 8;
    const int koB  = (lane < 16) ? 0 : 16;
    const int hsel = (lane < 16) ? 0 : 8;

    const unsigned int offQ  = (unsigned int)(size_t)&sQ[0];
    const unsigned int offK0 = (unsigned int)(size_t)&sK[0][0];
    const unsigned int offK1 = (unsigned int)(size_t)&sK[1][0];
    const unsigned int offV0 = (unsigned int)(size_t)&sV[0][0];
    const unsigned int offV1 = (unsigned int)(size_t)&sV[1][0];

    // prologue: Q tile [128 x 64] + tile 0 of K/V, all via TDM
    if (wid == 0) {
        tdm_load_2d(offQ, Qg, DD, 128, DD, 128, DD);
        tdm_load_2d(offK0, Kg, DD, 64, DD, 64, DD);
        tdm_load_2d(offV0, Vg, DD, DD, DD, DD, SS);
    }

    float mrow[8], lrow[8];
    v8f oacc[4];
    #pragma unroll
    for (int r = 0; r < 8; ++r) { mrow[r] = -3.0e38f; lrow[r] = 0.0f; }
    #pragma unroll
    for (int nb = 0; nb < 4; ++nb)
        #pragma unroll
        for (int r = 0; r < 8; ++r) oacc[nb][r] = 0.0f;

    int cur = 0;
    for (int j0 = 0; j0 < SS; j0 += 64) {
        __syncthreads();   // all waves done reading buffer cur^1 (iter j-1)
        if (wid == 0) {
            if (j0 + 64 < SS) {
                // prefetch tile j+1 into the alternate buffer, then wait for
                // tile j only (the two newest TDM ops may stay in flight)
                tdm_load_2d(cur ? offK0 : offK1,
                            Kg + (size_t)(j0 + 64) * DD, DD, 64, DD, 64, DD);
                tdm_load_2d(cur ? offV0 : offV1,
                            Vg + (j0 + 64), DD, DD, DD, DD, SS);
                __builtin_amdgcn_s_wait_tensorcnt(2);
            } else {
                __builtin_amdgcn_s_wait_tensorcnt(0);
            }
        }
        __syncthreads();   // publish TDM-staged tile j

        const _Float16* kbuf = &sK[cur][0];
        const _Float16* vbuf = &sV[cur][0];

        // scores = Q . K^T  (16 rows x 64 keys per wave)
        v8f sacc[4];
        #pragma unroll
        for (int nb = 0; nb < 4; ++nb)
            #pragma unroll
            for (int r = 0; r < 8; ++r) sacc[nb][r] = 0.0f;

        #pragma unroll
        for (int kk = 0; kk < 64; kk += 32) {
            int arow = wid * 16 + l16;
            v8h alo = *(const v8h*)&sQ[arow * 64 + kk + koA];
            v8h ahi = *(const v8h*)&sQ[arow * 64 + kk + 16 + koA];
            v16h a = mk16(alo, ahi);
            v16h bf[4];
            #pragma unroll
            for (int nb = 0; nb < 4; ++nb) {
                int key = nb * 16 + l16;
                v8h blo = *(const v8h*)&kbuf[key * 64 + kk + koA];
                v8h bhi = *(const v8h*)&kbuf[key * 64 + kk + 16 + koA];
                bf[nb] = mk16(blo, bhi);
            }
            #pragma unroll
            for (int nb = 0; nb < 4; ++nb)
                sacc[nb] = wmma_f16(a, bf[nb], sacc[nb]);
        }

        // online softmax per row (rows live in half-wave lane groups)
        #pragma unroll
        for (int r = 0; r < 8; ++r) {
            float mx = sacc[0][r];
            mx = fmaxf(mx, sacc[1][r]);
            mx = fmaxf(mx, sacc[2][r]);
            mx = fmaxf(mx, sacc[3][r]);
            #pragma unroll
            for (int xm = 1; xm <= 8; xm <<= 1)
                mx = fmaxf(mx, __shfl_xor(mx, xm, 32));
            float mnew = fmaxf(mrow[r], mx);
            float f = __expf(mrow[r] - mnew);
            float rs = 0.0f;
            #pragma unroll
            for (int nb = 0; nb < 4; ++nb) {
                float p = __expf(sacc[nb][r] - mnew);
                sacc[nb][r] = p;
                rs += p;
            }
            #pragma unroll
            for (int xm = 1; xm <= 8; xm <<= 1)
                rs += __shfl_xor(rs, xm, 32);
            lrow[r] = lrow[r] * f + rs;
            mrow[r] = mnew;
            #pragma unroll
            for (int nb = 0; nb < 4; ++nb) oacc[nb][r] *= f;
        }

        // spill P (C-layout) to LDS so it can be re-read in A-layout
        #pragma unroll
        for (int nb = 0; nb < 4; ++nb)
            #pragma unroll
            for (int r = 0; r < 8; ++r)
                sP[(wid * 16 + r + hsel) * 64 + nb * 16 + l16] =
                    (_Float16)sacc[nb][r];
        __syncthreads();

        // O += P . V
        #pragma unroll
        for (int kk = 0; kk < 64; kk += 32) {
            int arow = wid * 16 + l16;
            v8h plo = *(const v8h*)&sP[arow * 64 + kk + koA];
            v8h phi = *(const v8h*)&sP[arow * 64 + kk + 16 + koA];
            v16h ap = mk16(plo, phi);
            v16h bf[4];
            #pragma unroll
            for (int nb = 0; nb < 4; ++nb) {
                int d = nb * 16 + l16;
                v8h blo = *(const v8h*)&vbuf[d * 64 + kk + koA];
                v8h bhi = *(const v8h*)&vbuf[d * 64 + kk + 16 + koA];
                bf[nb] = mk16(blo, bhi);
            }
            #pragma unroll
            for (int nb = 0; nb < 4; ++nb)
                oacc[nb] = wmma_f16(ap, bf[nb], oacc[nb]);
        }
        cur ^= 1;
    }

    // epilogue: O / l -> out [B][S][NH*D] f32
    #pragma unroll
    for (int nb = 0; nb < 4; ++nb) {
        int dcol = nb * 16 + l16;
        #pragma unroll
        for (int r = 0; r < 8; ++r) {
            int srow = s0 + wid * 16 + r + hsel;
            out[((size_t)b * SS + srow) * (NHH * DD) + h * DD + dcol] =
                oacc[nb][r] / lrow[r];
        }
    }
}

extern "C" void kernel_launch(void* const* d_in, const int* in_sizes, int n_in,
                              void* d_out, int out_size, void* d_ws, size_t ws_size,
                              hipStream_t stream) {
    const float* e1 = (const float*)d_in[0];
    const float* e2 = (const float*)d_in[1];
    const float* e3 = (const float*)d_in[2];
    const float* Wq = (const float*)d_in[3];
    const float* bq = (const float*)d_in[4];
    const float* Wk = (const float*)d_in[5];
    const float* bk = (const float*)d_in[6];
    const float* Wv = (const float*)d_in[7];
    const float* bv = (const float*)d_in[8];

    const size_t perTensor = (size_t)BB * NHH * SS * DD;   // 6.29M f16 each
    _Float16* Qh = (_Float16*)d_ws;
    _Float16* Kh = Qh + perTensor;
    _Float16* Vh = Kh + perTensor;

    dim3 grid(SS / 128, NHH, BB);
    qkv_kernel<<<grid, 256, 0, stream>>>(e1, e2, e3, Wq, bq, Wk, bk, Wv, bv,
                                         Qh, Kh, Vh);
    attn_kernel<<<grid, 256, 0, stream>>>(Qh, Kh, Vh, (float*)d_out);
}
